// Relation_72112500899921
// MI455X (gfx1250) — matrix operand order repaired and checked by reading, not compiled, not run on hardware
//
#include <hip/hip_runtime.h>
#include <hip/hip_bf16.h>
#include <math.h>
#include <stdint.h>

// ---------------------------------------------------------------------------
// Relation module == multi-head attention after hoisting the grouped V-
// projection in front of the attention-weighted sum (linear commutes with the
// convex combination; softmax rows sum to 1 so Wv_b folds into V):
//   Q  = (roi @ Wq^T + bq) * (log2e/sqrt(DG))   [N,F] bf16   (scale pre-folded)
//   K  = ref @ Wk^T + bk                        [M,F] bf16
//   Vt = (ref @ Wv^T + bv)^T                    [F,M] bf16, TRANSPOSED
//   out[n, g*DG+d] = softmax2_m(Q_g K_g^T) @ V_g      (f32, base-2 softmax)
// ---------------------------------------------------------------------------

typedef __attribute__((ext_vector_type(16))) __bf16 v16bf;
typedef __attribute__((ext_vector_type(8)))  __bf16 v8bf;
typedef __attribute__((ext_vector_type(8)))  float  v8f;

#define FDIM 2048
#define NROI 2048
#define MREF 2048
#define GRP  16
#define DG   128
#define MT   64            // attention M tile
#define NT   (MREF / MT)   // 32 iterations

// bare v_exp_f32 (base-2 exponential)
#define EXP2F(x) __builtin_amdgcn_exp2f(x)

static __device__ __forceinline__ v16bf cat8(v8bf lo, v8bf hi) {
  return __builtin_shufflevector(lo, hi, 0,1,2,3,4,5,6,7,8,9,10,11,12,13,14,15);
}

static __device__ __forceinline__ v8bf pack8(float4 a, float4 b) {
  v8bf r;
  r[0] = (__bf16)a.x; r[1] = (__bf16)a.y; r[2] = (__bf16)a.z; r[3] = (__bf16)a.w;
  r[4] = (__bf16)b.x; r[5] = (__bf16)b.y; r[6] = (__bf16)b.z; r[7] = (__bf16)b.w;
  return r;
}

static __device__ __forceinline__ float red_max16(float v) {
  v = fmaxf(v, __shfl_xor(v, 1));
  v = fmaxf(v, __shfl_xor(v, 2));
  v = fmaxf(v, __shfl_xor(v, 4));
  v = fmaxf(v, __shfl_xor(v, 8));
  return v;   // reduced across the 16 lanes of this half-wave
}
static __device__ __forceinline__ float red_sum16(float v) {
  v += __shfl_xor(v, 1);
  v += __shfl_xor(v, 2);
  v += __shfl_xor(v, 4);
  v += __shfl_xor(v, 8);
  return v;
}

// CDNA5 async DMA: copy 16 bytes global -> LDS per lane (ASYNCcnt tracked).
// Generic pointers to __shared__ carry the LDS byte offset in their low 32
// bits (LDS aperture tag lives in [63:32]).
static __device__ __forceinline__ void async_copy16(void* lds_dst, const void* gsrc) {
  unsigned loff = (unsigned)(uintptr_t)lds_dst;
  asm volatile("global_load_async_to_lds_b128 %0, %1, off"
               :: "v"(loff), "v"(gsrc) : "memory");
}
static __device__ __forceinline__ void async_wait0() {
  asm volatile("s_wait_asynccnt 0" ::: "memory");
}

// ---------------------------------------------------------------------------
// C = (A @ W^T + bias) * oscale  -> bf16.
//   TR=false: C[r][c] row-major [rows,F].   TR=true: C[c][r] transposed [F,rows].
// block = 256 thr (8 waves); tile = 128 rows x 128 cols; k-step = 64.
// ---------------------------------------------------------------------------
template<bool TR>
__global__ __launch_bounds__(256) void gemm_xWT_bias_bf16(
    const float* __restrict__ A, const float* __restrict__ W,
    const float* __restrict__ bias, __bf16* __restrict__ C, float oscale)
{
  __shared__ __align__(32) __bf16 ldsB[128][64];   // [out-col][k]  16 KB

  const int tid  = threadIdx.x;
  const int wave = tid >> 5;
  const int lane = tid & 31;
  const int half = lane >> 4;   // 0: lanes 0-15, 1: lanes 16-31
  const int l16  = lane & 15;
  const int colBase = blockIdx.x * 128;
  const int rowBase = blockIdx.y * 128;
  const int row = rowBase + wave * 16 + l16;      // this lane's A row

  v8f acc[8] = {};              // 8 col-tiles of 16x16 f32

  const int wr = tid >> 1;          // 0..127  staged W row (out col)
  const int wc = (tid & 1) << 5;    // 0 or 32

  for (int k0 = 0; k0 < FDIM; k0 += 64) {
    { // cooperative W stage: 128x64 f32 -> bf16 LDS (32 elems / thread)
      const float* src = W + (size_t)(colBase + wr) * FDIM + k0 + wc;
#pragma unroll
      for (int q8 = 0; q8 < 4; ++q8) {
        float4 w0 = *(const float4*)(src + q8 * 8);
        float4 w1 = *(const float4*)(src + q8 * 8 + 4);
        *(v8bf*)(&ldsB[wr][wc + q8 * 8]) = pack8(w0, w1);
      }
    }
    __syncthreads();

    // Two A tiles (16x32 each) in contiguous k order (same as B): half 0 holds K{0..15},
    // half 1 holds K{16..31} -> one contiguous 16-float chunk per lane.
    v16bf a[2];
    {
      const float* ap = A + (size_t)row * FDIM + k0;
#pragma unroll
      for (int ks = 0; ks < 2; ++ks) {
        const float* p = ap + ks * 32;
        float4 f0 = *(const float4*)(p + half * 16);
        float4 f1 = *(const float4*)(p + half * 16 + 4);
        float4 f2 = *(const float4*)(p + half * 16 + 8);
        float4 f3 = *(const float4*)(p + half * 16 + 12);
        a[ks] = cat8(pack8(f0, f1), pack8(f2, f3));
      }
    }

#pragma unroll
    for (int ct = 0; ct < 8; ++ct) {
#pragma unroll
      for (int ks = 0; ks < 2; ++ks) {
        // B (32x16): lane N=l16 -> out-col row of ldsB; K = e + 16*half
        v16bf b = *(const v16bf*)(&ldsB[ct * 16 + l16][ks * 32 + half * 16]);
        acc[ct] = __builtin_amdgcn_wmma_f32_16x16x32_bf16(
            false, a[ks], false, b, (short)0, acc[ct], false, false);
      }
    }
    __syncthreads();
  }

#pragma unroll
  for (int ct = 0; ct < 8; ++ct) {
    const int col = colBase + ct * 16 + l16;
    const float bv = bias[col];
    if (TR) {
      // transposed: lane's 8 values are 8 consecutive rows -> one 16B store
      v8bf pk;
#pragma unroll
      for (int j = 0; j < 8; ++j) pk[j] = (__bf16)((acc[ct][j] + bv) * oscale);
      *(v8bf*)(C + (size_t)col * MREF + rowBase + wave * 16 + half * 8) = pk;
    } else {
#pragma unroll
      for (int j = 0; j < 8; ++j) {
        const int r = rowBase + wave * 16 + half * 8 + j;   // C/D layout row
        C[(size_t)r * FDIM + col] = (__bf16)((acc[ct][j] + bv) * oscale);
      }
    }
  }
}

// ---------------------------------------------------------------------------
// Flash attention, base-2 softmax (scale folded into Q at GEMM time).
// block = (head g, 128 q rows), 8 waves x 16 q rows. Q (4 A-tiles) and
// O (8 f32 C-tiles) live in registers. K / V^T tiles are double-buffered in
// LDS and fetched with async global->LDS DMA, overlapped with compute.
// ---------------------------------------------------------------------------
__global__ __launch_bounds__(256) void attn_flash_bf16(
    const __bf16* __restrict__ Qb, const __bf16* __restrict__ Kb,
    const __bf16* __restrict__ Vt, float* __restrict__ out)
{
  __shared__ __align__(32) __bf16 ldsK [2][MT][DG];   // [buf][m][d]   32 KB
  __shared__ __align__(32) __bf16 ldsVt[2][DG][MT];   // [buf][d][m]   32 KB
  __shared__ __align__(32) __bf16 ldsP [8][16][MT];   // [wave][q][m]  16 KB

  const int tid  = threadIdx.x;
  const int wave = tid >> 5;
  const int lane = tid & 31;
  const int half = lane >> 4;
  const int l16  = lane & 15;
  const int g     = blockIdx.x;          // head
  const int qBase = blockIdx.y * 128;    // q row block

  // issue async stage of tile t into buffer b (8 x 16B chunks per thread)
  auto stage = [&](int t, int b) {
    const int m0 = t * MT;
#pragma unroll
    for (int i = 0; i < 4; ++i) {
      const int c  = tid + i * 256;       // 1024 chunks of 8 bf16
      const int r  = c >> 4;              // m 0..63
      const int cc = (c & 15) << 3;       // d 0..120
      async_copy16(&ldsK[b][r][cc], Kb + (size_t)(m0 + r) * FDIM + g * DG + cc);
    }
#pragma unroll
    for (int i = 0; i < 4; ++i) {
      const int c  = tid + i * 256;
      const int d  = c >> 3;              // d 0..127
      const int mm = (c & 7) << 3;        // m 0..56
      async_copy16(&ldsVt[b][d][mm], Vt + (size_t)(g * DG + d) * MREF + m0 + mm);
    }
  };

  // ---- Q block into registers: 4 A-tiles (16 rows x 128 d) ----
  v16bf qa[4];
  {
    const __bf16* qp = Qb + (size_t)(qBase + wave * 16 + l16) * FDIM + g * DG;
#pragma unroll
    for (int ks = 0; ks < 4; ++ks) {
      v8bf lo = *(const v8bf*)(qp + ks * 32 + half * 16);
      v8bf hi = *(const v8bf*)(qp + ks * 32 + half * 16 + 8);
      qa[ks] = cat8(lo, hi);
    }
  }

  float m_i[8], l_i[8];
  v8f o[8] = {};
#pragma unroll
  for (int j = 0; j < 8; ++j) { m_i[j] = -INFINITY; l_i[j] = 0.0f; }

  stage(0, 0);   // prologue fetch

  for (int t = 0; t < NT; ++t) {
    const int buf = t & 1;
    async_wait0();       // this wave's chunks of tile t have landed
    __syncthreads();     // everyone's chunks landed; prev-iter reads all done
    if (t + 1 < NT) stage(t + 1, buf ^ 1);   // overlap DMA with compute

    // ---- S = Q K^T (log2-scaled; scale folded into Q) : 4 m-tiles ----
    v8f s[4] = {};
#pragma unroll
    for (int mt = 0; mt < 4; ++mt) {
#pragma unroll
      for (int ks = 0; ks < 4; ++ks) {
        // B(32x16): N = m (ldsK row), K = d contiguous
        v16bf b = *(const v16bf*)(&ldsK[buf][mt * 16 + l16][ks * 32 + half * 16]);
        s[mt] = __builtin_amdgcn_wmma_f32_16x16x32_bf16(
            false, qa[ks], false, b, (short)0, s[mt], false, false);
      }
    }

    // ---- online softmax update (base 2) ----
    float alpha[8];
#pragma unroll
    for (int j = 0; j < 8; ++j) {
      float r = fmaxf(fmaxf(s[0][j], s[1][j]), fmaxf(s[2][j], s[3][j]));
      r = red_max16(r);
      const float mnew = fmaxf(m_i[j], r);
      alpha[j] = EXP2F(m_i[j] - mnew);
      m_i[j] = mnew;
      l_i[j] *= alpha[j];
    }
#pragma unroll
    for (int dt = 0; dt < 8; ++dt)
#pragma unroll
      for (int j = 0; j < 8; ++j) o[dt][j] *= alpha[j];

    // ---- P = exp2(S - m), row sums, spill P to LDS for A-layout reload ----
#pragma unroll
    for (int mt = 0; mt < 4; ++mt) {
#pragma unroll
      for (int j = 0; j < 8; ++j) {
        const float p = EXP2F(s[mt][j] - m_i[j]);
        s[mt][j] = p;
        ldsP[wave][half * 8 + j][mt * 16 + l16] = (__bf16)p;
      }
    }
#pragma unroll
    for (int j = 0; j < 8; ++j) {
      float rs = s[0][j] + s[1][j] + s[2][j] + s[3][j];
      l_i[j] += red_sum16(rs);
    }
    // same-wave LDS RAW: P stores above feed the A-layout loads below
    asm volatile("s_wait_dscnt 0" ::: "memory");

    // ---- O += P @ V ----
    v16bf pa[2];
    {
      const __bf16* pp = &ldsP[wave][l16][0];
#pragma unroll
      for (int ks2 = 0; ks2 < 2; ++ks2) {
        v8bf lo = *(const v8bf*)(pp + ks2 * 32 + half * 16);
        v8bf hi = *(const v8bf*)(pp + ks2 * 32 + half * 16 + 8);
        pa[ks2] = cat8(lo, hi);
      }
    }
#pragma unroll
    for (int dt = 0; dt < 8; ++dt) {
#pragma unroll
      for (int ks2 = 0; ks2 < 2; ++ks2) {
        // B(32x16): N = d (ldsVt row), K = m contiguous
        v16bf b = *(const v16bf*)(&ldsVt[buf][dt * 16 + l16][ks2 * 32 + half * 16]);
        o[dt] = __builtin_amdgcn_wmma_f32_16x16x32_bf16(
            false, pa[ks2], false, b, (short)0, o[dt], false, false);
      }
    }
  }

  // ---- epilogue: O / l -> out (f32) ----
  float inv[8];
#pragma unroll
  for (int j = 0; j < 8; ++j) inv[j] = 1.0f / l_i[j];
#pragma unroll
  for (int dt = 0; dt < 8; ++dt) {
#pragma unroll
    for (int j = 0; j < 8; ++j) {
      const int r = qBase + wave * 16 + half * 8 + j;
      out[(size_t)r * FDIM + g * DG + dt * 16 + l16] = o[dt][j] * inv[j];
    }
  }
}

// ---------------------------------------------------------------------------
extern "C" void kernel_launch(void* const* d_in, const int* in_sizes, int n_in,
                              void* d_out, int out_size, void* d_ws, size_t ws_size,
                              hipStream_t stream) {
  (void)in_sizes; (void)n_in; (void)out_size; (void)ws_size;
  const float* roi  = (const float*)d_in[0];
  const float* ref  = (const float*)d_in[1];
  const float* Wq_w = (const float*)d_in[2];
  const float* Wq_b = (const float*)d_in[3];
  const float* Wk_w = (const float*)d_in[4];
  const float* Wk_b = (const float*)d_in[5];
  const float* Wv_w = (const float*)d_in[6];
  const float* Wv_b = (const float*)d_in[7];
  float* out = (float*)d_out;

  __bf16* Qb = (__bf16*)d_ws;                       // 8 MB, [N][F]
  __bf16* Kb = Qb + (size_t)NROI * FDIM;            // 8 MB, [M][F]
  __bf16* Vt = Kb + (size_t)MREF * FDIM;            // 8 MB, [F][M] transposed

  // log2(e) / sqrt(DG): base-2 softmax scale folded into Q
  const float qscale = 1.4426950408889634f / sqrtf((float)DG);

  dim3 gdim(FDIM / 128, NROI / 128);                // 16 x 16
  gemm_xWT_bias_bf16<false><<<gdim, 256, 0, stream>>>(roi, Wq_w, Wq_b, Qb, qscale);
  gemm_xWT_bias_bf16<false><<<gdim, 256, 0, stream>>>(ref, Wk_w, Wk_b, Kb, 1.0f);
  gemm_xWT_bias_bf16<true ><<<gdim, 256, 0, stream>>>(ref, Wv_w, Wv_b, Vt, 1.0f);

  attn_flash_bf16<<<dim3(GRP, NROI / 128), 256, 0, stream>>>(Qb, Kb, Vt, out);
}
